// ObjectImplicitNetworkGrid_19370302505667
// MI455X (gfx1250) — matrix operand order repaired and checked by reading, not compiled, not run on hardware
//
#include <hip/hip_runtime.h>
#include <hip/hip_fp16.h>
#include <hip/hip_bf16.h>

typedef __attribute__((ext_vector_type(16))) _Float16 v16h;
typedef __attribute__((ext_vector_type(8)))  float    v8f;

// ---------------- model constants (from reference) ----------------
#define NLAYERS 9
constexpr int IN_D [NLAYERS] = {35,256,256,256,291,256,256,256,256};
constexpr int OUT_D[NLAYERS] = {256,256,256,256,256,256,256,256,261};
constexpr int KPAD [NLAYERS] = {64,256,256,256,320,256,256,256,256};   // K padded to mult of 32
constexpr int NPAD [NLAYERS] = {256,256,256,256,256,256,256,256,272}; // N padded to mult of 16
constexpr size_t WOFF[NLAYERS] = {0,16384,81920,147456,212992,294912,360448,425984,491520};
constexpr size_t WTOTAL = 561152;      // total f16 weight elements
constexpr int BOFF[NLAYERS] = {0,256,512,768,1024,1280,1536,1792,2048};
constexpr int BTOTAL = 2320;           // total padded bias floats
constexpr size_t WS_BYTES = WTOTAL*2 + (size_t)BTOTAL*4;
constexpr int TOTAL_NEURONS = 2309;    // 8*256 + 261

// hash grid metadata (precomputed from torch-ngp HashEncoder config)
constexpr int RES_A[16] = {16,23,31,43,59,81,112,154,213,295,407,562,777,1073,1483,2048};
constexpr int OFF_A[16] = {0,4096,16264,46056,125568,330952,855240,1379528,
                           1903816,2428104,2952392,3476680,4000968,4525256,5049544,5573832};
// levels 0..4 dense (res^3 <= 2^19), 5..15 hashed

struct MlpPtrs { const float* v[NLAYERS]; const float* g[NLAYERS]; const float* b[NLAYERS]; };

// ---------------- weight preparation ----------------
__global__ void zero_ws_kernel(uint32_t* ws, int ndw) {
  int i = blockIdx.x * blockDim.x + threadIdx.x;
  int stride = gridDim.x * blockDim.x;
  for (; i < ndw; i += stride) ws[i] = 0u;
}

// one wave32 per output neuron: compute g/||v||, write transposed padded f16 weights
__global__ __launch_bounds__(32) void prep_weights_kernel(MlpPtrs P, _Float16* wt, float* bias) {
  int nid = blockIdx.x;
  int l = 0, base = 0;
  while (l < NLAYERS - 1 && nid >= base + OUT_D[l]) { base += OUT_D[l]; ++l; }
  int o = nid - base;
  if (o >= OUT_D[l]) return;
  const int ind = IN_D[l];
  const float* vrow = P.v[l] + (size_t)o * ind;
  int lane = threadIdx.x;
  float ss = 0.f;
  for (int i = lane; i < ind; i += 32) { float t = vrow[i]; ss += t * t; }
  for (int m = 16; m; m >>= 1) ss += __shfl_xor(ss, m, 32);
  float scale = P.g[l][o] * rsqrtf(ss);
  if (l == 4) scale *= 0.70710678118f;   // fold the skip-layer 1/sqrt(2) into W4
  _Float16* w = wt + WOFF[l];
  const int np = NPAD[l];
  for (int i = lane; i < ind; i += 32)
    w[(size_t)i * np + o] = (_Float16)(vrow[i] * scale);
  if (lane == 0) bias[BOFF[l] + o] = P.b[l][o];
}

// ---------------- hash-grid encoder (8 levels per call, fully unrolled) ----------------
template<int L0>
__device__ __forceinline__ void encode8(const float* __restrict__ table,
                                        float ux, float uy, float uz, _Float16* dst) {
#pragma unroll
  for (int li = 0; li < 8; ++li) {
    const int l = L0 + li;
    const int res = RES_A[l];
    const float rf = (float)(res - 1);
    float px = ux * rf, py = uy * rf, pz = uz * rf;
    float fx = floorf(px), fy = floorf(py), fz = floorf(pz);
    float tx = px - fx, ty = py - fy, tz = pz - fz;
    unsigned xi = (unsigned)fx, yi = (unsigned)fy, zi = (unsigned)fz;
    const unsigned rm = (unsigned)(res - 1);
    float a0 = 0.f, a1 = 0.f;
#pragma unroll
    for (int c = 0; c < 8; ++c) {
      unsigned bx = c & 1, by = (c >> 1) & 1, bz = (c >> 2) & 1;
      unsigned cx = xi + bx; if (cx > rm) cx = rm;
      unsigned cy = yi + by; if (cy > rm) cy = rm;
      unsigned cz = zi + bz; if (cz > rm) cz = rm;
      float w = (bx ? tx : 1.f - tx) * (by ? ty : 1.f - ty) * (bz ? tz : 1.f - tz);
      unsigned idx;
      if (l < 5) idx = cx + cy * (unsigned)res + cz * (unsigned)(res * res);
      else       idx = (cx ^ (cy * 2654435761u) ^ (cz * 805459861u)) & ((1u << 19) - 1u);
      float2 tv = ((const float2*)table)[(size_t)OFF_A[l] + idx];
      a0 += w * tv.x; a1 += w * tv.y;
    }
    dst[2 * li]     = (_Float16)a0;
    dst[2 * li + 1] = (_Float16)a1;
  }
}

// ---------------- column-parallel MLP layer ----------------
// Wave owns TPW col-tiles starting at tile0 and iterates all 8 row-tiles (128 rows).
// B (weights, TPW tiles = 16*TPW strided f16 loads per lane) is double-buffered ACROSS the runtime k-loop:
// bc is live across the backedge, so the scheduler cannot sink its loads to their uses;
// the global-load wait lands once per k-step, amortized over 16 WMMAs.
// A fragments stream from LDS through a 2-buffer rotation over the row-tile loop so the
// LDS latency of row-tile rt+1 overlaps the two WMMAs of row-tile rt.
template<int TPW, int NP, bool SKIPL, bool LAST>
__device__ __forceinline__ void mlp_layer_col(
    int lane, int tile0, int Kp,
    const _Float16* __restrict__ wt, const float* __restrict__ bias,
    const _Float16* hin, int hstride, const _Float16* inpb,
    _Float16* hout, float* __restrict__ out, int growBase)
{
  const int aRow = lane & 15;
  const int aHi  = lane >> 4;

  v8f acc[8][TPW];
#pragma unroll
  for (int rt = 0; rt < 8; ++rt)
#pragma unroll
    for (int j = 0; j < TPW; ++j)
#pragma unroll
      for (int r = 0; r < 8; ++r) acc[rt][j][r] = 0.f;

  union BU { v16h h; uint4 q[2]; };
  union AU { v16h h; uint4 q[2]; };
  BU bc[TPW], bn[TPW];

  const _Float16* brow0 = wt + (size_t)(aHi * 8) * NP + tile0 * 16 + aRow;
#pragma unroll
  for (int j = 0; j < TPW; ++j) {
#pragma unroll
    for (int s = 0; s < 16; ++s)
      bc[j].h[s] = brow0[(size_t)((s & 7) + (s >> 3) * 16) * NP + j * 16];
  }

  for (int k0 = 0; k0 < Kp; k0 += 32) {
    const bool more = (k0 + 32 < Kp);
    const _Float16* brown = wt + (size_t)(k0 + 32 + aHi * 8) * NP + tile0 * 16 + aRow;
    if (more) {
#pragma unroll
      for (int j = 0; j < TPW; ++j) {            // next k-step's B, consumed next iter
#pragma unroll
        for (int s = 0; s < 16; ++s)
          bn[j].h[s] = brown[(size_t)((s & 7) + (s >> 3) * 16) * NP + j * 16];
      }
      __builtin_prefetch(brown + 32 * NP, 0, 3); // warm k0+64 slab in near cache
    }

    const _Float16* abase0;
    int astr;
    if (SKIPL && k0 >= 256) { abase0 = inpb + (k0 - 256); astr = 64; }
    else                    { abase0 = hin + k0;          astr = hstride; }
    const _Float16* abase = abase0 + aRow * astr + aHi * 8;

    AU Aa[2];
    Aa[0].q[0] = *(const uint4*)(abase);                 // preload row-tile 0
    Aa[0].q[1] = *(const uint4*)(abase + 16);
#pragma unroll
    for (int rt = 0; rt < 8; ++rt) {
      if (rt + 1 < 8) {                                  // A for rt+1 ahead of rt's WMMAs
        const _Float16* abn = abase + (rt + 1) * 16 * astr;
        Aa[(rt + 1) & 1].q[0] = *(const uint4*)(abn);
        Aa[(rt + 1) & 1].q[1] = *(const uint4*)(abn + 16);
      }
#pragma unroll
      for (int j = 0; j < TPW; ++j)
        acc[rt][j] = __builtin_amdgcn_wmma_f32_16x16x32_f16(
            false, Aa[rt & 1].h, false, bc[j].h, (short)0, acc[rt][j], false, false);
    }
    if (more) {
#pragma unroll
      for (int j = 0; j < TPW; ++j) bc[j] = bn[j];  // v_movs; co-execute with XDL
    }
  }

  // Epilogue: C/D layout — VGPR r: lanes0-15 -> M=r, lanes16-31 -> M=r+8; col = lane&15
  const int colLane = lane & 15;
#pragma unroll
  for (int rt = 0; rt < 8; ++rt) {
#pragma unroll
    for (int j = 0; j < TPW; ++j) {
      const int col = (tile0 + j) * 16 + colLane;
      const float bv = bias[col];
#pragma unroll
      for (int r = 0; r < 8; ++r) {
        float v = acc[rt][j][r] + bv;
        const int row = rt * 16 + aHi * 8 + r;
        if (!LAST) {
          float z = 100.f * v;               // softplus(100x)/100, numerically stable
          v = fmaxf(v, 0.f) + 0.01f * __logf(1.f + __expf(-fabsf(z)));
          hout[row * 256 + col] = (_Float16)v;
        } else {
          if (col < 261) out[(size_t)(growBase + row) * 261 + col] = v;
        }
      }
    }
  }
}

// Tail tile of the last layer (padded cols 256..271, valid 256..260): tiny row-parallel
// pass — each wave handles its own 16-row slab, K=256, 8 WMMAs total.
__device__ __forceinline__ void mlp_tail_tile(
    int lane, int m0,
    const _Float16* __restrict__ wt /* + 256 col offset */, const float* __restrict__ bias,
    const _Float16* hin, float* __restrict__ out, int growBase)
{
  const int aRow = lane & 15;
  const int aHi  = lane >> 4;
  v8f acc;
#pragma unroll
  for (int r = 0; r < 8; ++r) acc[r] = 0.f;

  for (int k0 = 0; k0 < 256; k0 += 32) {
    const _Float16* ab = hin + (m0 + aRow) * 256 + k0 + aHi * 8;
    union { v16h h; uint4 q[2]; } A, B;
    A.q[0] = *(const uint4*)(ab);
    A.q[1] = *(const uint4*)(ab + 16);
    const _Float16* brow = wt + (size_t)(k0 + aHi * 8) * 272 + aRow;
#pragma unroll
    for (int s = 0; s < 16; ++s)
      B.h[s] = brow[(size_t)((s & 7) + (s >> 3) * 16) * 272];
    acc = __builtin_amdgcn_wmma_f32_16x16x32_f16(
        false, A.h, false, B.h, (short)0, acc, false, false);
  }
  const int col = 256 + (lane & 15);
  if (col < 261) {
    const float bv = bias[col];
#pragma unroll
    for (int r = 0; r < 8; ++r) {
      const int row = m0 + (lane >> 4) * 8 + r;
      out[(size_t)(growBase + row) * 261 + col] = acc[r] + bv;
    }
  }
}

// ---------------- fused encoder + MLP ----------------
__global__ __launch_bounds__(256) __attribute__((amdgpu_waves_per_eu(1, 8)))
void fused_kernel(
    const float* __restrict__ xin, const float* __restrict__ table,
    const _Float16* __restrict__ wt, const float* __restrict__ bias,
    float* __restrict__ out)
{
  extern __shared__ char smem[];
  _Float16* inp = (_Float16*)smem;                    // [128][64]  padded MLP input
  _Float16* h0  = (_Float16*)(smem + 16384);          // [128][256] activations ping
  _Float16* h1  = (_Float16*)(smem + 16384 + 65536);  // [128][256] activations pong

  const int tid  = threadIdx.x;
  const int p    = tid >> 1;       // point within tile (2 threads/point)
  const int hsel = tid & 1;        // which 8 levels this thread encodes
  const int grow = blockIdx.x * 128 + p;

  const float x0 = xin[3 * grow + 0];
  const float x1 = xin[3 * grow + 1];
  const float x2 = xin[3 * grow + 2];
  const float inv = 1.0f / 1.5f;   // DIVIDE_FACTOR
  const float ux = (x0 * inv + 1.f) * 0.5f;
  const float uy = (x1 * inv + 1.f) * 0.5f;
  const float uz = (x2 * inv + 1.f) * 0.5f;

  _Float16* irow = inp + p * 64;
  if (hsel == 0) {
    irow[0] = (_Float16)x0; irow[1] = (_Float16)x1; irow[2] = (_Float16)x2;
    encode8<0>(table, ux, uy, uz, irow + 3);          // cols 3..18
  } else {
    for (int c = 35; c < 64; ++c) irow[c] = (_Float16)0.f;  // zero K-padding
    encode8<8>(table, ux, uy, uz, irow + 3 + 16);     // cols 19..34
  }
  __syncthreads();

  const int lane  = tid & 31;
  const int wid   = tid >> 5;
  const int tile0 = wid * 2;       // each wave owns 2 col-tiles (32 output columns)
  const int base  = blockIdx.x * 128;

  mlp_layer_col<2,256,false,false>(lane, tile0,  64, wt + WOFF[0], bias + BOFF[0], inp,  64, nullptr, h0, nullptr, 0);
  __syncthreads();
  mlp_layer_col<2,256,false,false>(lane, tile0, 256, wt + WOFF[1], bias + BOFF[1], h0, 256, nullptr, h1, nullptr, 0);
  __syncthreads();
  mlp_layer_col<2,256,false,false>(lane, tile0, 256, wt + WOFF[2], bias + BOFF[2], h1, 256, nullptr, h0, nullptr, 0);
  __syncthreads();
  mlp_layer_col<2,256,false,false>(lane, tile0, 256, wt + WOFF[3], bias + BOFF[3], h0, 256, nullptr, h1, nullptr, 0);
  __syncthreads();
  mlp_layer_col<2,256,true ,false>(lane, tile0, 320, wt + WOFF[4], bias + BOFF[4], h1, 256, inp,     h0, nullptr, 0);
  __syncthreads();
  mlp_layer_col<2,256,false,false>(lane, tile0, 256, wt + WOFF[5], bias + BOFF[5], h0, 256, nullptr, h1, nullptr, 0);
  __syncthreads();
  mlp_layer_col<2,256,false,false>(lane, tile0, 256, wt + WOFF[6], bias + BOFF[6], h1, 256, nullptr, h0, nullptr, 0);
  __syncthreads();
  mlp_layer_col<2,256,false,false>(lane, tile0, 256, wt + WOFF[7], bias + BOFF[7], h0, 256, nullptr, h1, nullptr, 0);
  __syncthreads();
  mlp_layer_col<2,272,false,true >(lane, tile0, 256, wt + WOFF[8], bias + BOFF[8], h1, 256, nullptr, nullptr, out, base);
  mlp_tail_tile(lane, wid * 16, wt + WOFF[8] + 256, bias + BOFF[8], h1, out, base);
}

// ---------------- host launch ----------------
extern "C" void kernel_launch(void* const* d_in, const int* in_sizes, int n_in,
                              void* d_out, int out_size, void* d_ws, size_t ws_size,
                              hipStream_t stream)
{
  const float* x     = (const float*)d_in[0];
  const float* table = (const float*)d_in[1];

  MlpPtrs P;
  if (n_in >= 2 + 3 * NLAYERS) {
    // leaf-flattened: d_in[2..10]=vs, [11..19]=gs, [20..28]=bs
    for (int l = 0; l < NLAYERS; ++l) {
      P.v[l] = (const float*)d_in[2 + l];
      P.g[l] = (const float*)d_in[2 + NLAYERS + l];
      P.b[l] = (const float*)d_in[2 + 2 * NLAYERS + l];
    }
  } else {
    // concatenated fallback: d_in[2]=vs cat, d_in[3]=gs cat, d_in[4]=bs cat
    const float* vcat = (const float*)d_in[2];
    const float* gcat = (const float*)d_in[3];
    const float* bcat = (const float*)d_in[4];
    size_t vo = 0, go = 0;
    for (int l = 0; l < NLAYERS; ++l) {
      P.v[l] = vcat + vo; vo += (size_t)OUT_D[l] * IN_D[l];
      P.g[l] = gcat + go;
      P.b[l] = bcat + go; go += OUT_D[l];
    }
  }

  if (ws_size < WS_BYTES) return;
  _Float16* wt = (_Float16*)d_ws;
  float* bias  = (float*)((char*)d_ws + WTOTAL * 2);

  zero_ws_kernel<<<512, 256, 0, stream>>>((uint32_t*)d_ws, (int)(WS_BYTES / 4));
  prep_weights_kernel<<<TOTAL_NEURONS, 32, 0, stream>>>(P, wt, bias);

  const int smemBytes = 16384 + 2 * 65536;  // 144 KB dynamic LDS (inp + ping/pong)
  fused_kernel<<<1024, 256, smemBytes, stream>>>(x, table, wt, bias, (float*)d_out);
}
